// GNNEncoder_24464133718256
// MI455X (gfx1250) — matrix operand order repaired and checked by reading, not compiled, not run on hardware
//
#include <hip/hip_runtime.h>

typedef __attribute__((ext_vector_type(16))) _Float16 v16h;
typedef __attribute__((ext_vector_type(8)))  _Float16 v8h;
typedef __attribute__((ext_vector_type(2)))  _Float16 v2h;
typedef __attribute__((ext_vector_type(8)))  float    v8f;

union V16 { v16h v; v8h h[2]; };

// CDNA5 split-counter LDS fence: order same-wave cross-lane LDS store->load.
__device__ __forceinline__ void lds_fence() {
    asm volatile("s_wait_dscnt 0" ::: "memory");
}

__device__ __forceinline__ v8f wmma_f16(v16h a, v16h b, v8f c) {
    // D = A(16x32 f16) * B(32x16 f16) + C(16x16 f32)
    return __builtin_amdgcn_wmma_f32_16x16x32_f16(
        false, a, false, b, (short)0, c, false, false);
}

#define NWAVE       4
#define BLK         (NWAVE * 32)
#define S_TOTAL     2048
#define SCHUNKS     32
#define S_PER_BLOCK (S_TOTAL / SCHUNKS)    // 64
#define S_PER_WAVE  (S_PER_BLOCK / NWAVE)  // 16

__global__ void zero_out_kernel(float* __restrict__ p, int n) {
    int i = blockIdx.x * blockDim.x + threadIdx.x;
    if (i < n) p[i] = 0.0f;
}

__global__ __launch_bounds__(BLK)
void gnn_encoder_kernel(const float* __restrict__ x,      // [16,2048,64]
                        const float* __restrict__ adj,    // [64,64]
                        const float* __restrict__ fc1_w,  // [32,1]
                        const float* __restrict__ fc1_b,  // [32]
                        const float* __restrict__ fc2_w,  // [32,64]
                        const float* __restrict__ fc2_b,  // [32]
                        float* __restrict__ out)          // [16,2048]
{
    // Padded strides (multiples of 16B) to avoid LDS bank conflicts.
    __shared__ __align__(16) _Float16 adjn16[64][72];        // adj_norm, f16 row-major
    __shared__ __align__(16) _Float16 w2t[32][72];           // fc2_w rows: [out_f][in 0..63]
    __shared__ float w1s[32], b1s[32], b2s[32];
    __shared__ __align__(16) _Float16 xh[NWAVE][64];         // x row, f16
    __shared__ __align__(16) _Float16 hcm[NWAVE][32][72];    // h column-major [feat][var]
    __shared__ __align__(16) _Float16 aggrm[NWAVE][64][40];  // agg row-major  [var][feat]

    const int tid  = threadIdx.x;
    const int lane = tid & 31;
    const int wv   = tid >> 5;
    const int hf   = lane >> 4;     // lane half (0/1)
    const int l15  = lane & 15;
    const int b    = blockIdx.x;
    const int s0   = blockIdx.y * S_PER_BLOCK + wv * S_PER_WAVE;

    // ---------------- one-time block setup ----------------
    if (tid < 64) {
        float sum = 0.0f;
        for (int j = 0; j < 64; ++j) sum += adj[tid * 64 + j];
        float inv = 1.0f / (sum + 1e-8f);
        for (int j = 0; j < 64; ++j)
            adjn16[tid][j] = (_Float16)(adj[tid * 64 + j] * inv);
    }
    for (int e = tid; e < 32 * 64; e += BLK) {
        int n = e >> 6, k = e & 63;
        w2t[n][k] = (_Float16)fc2_w[n * 64 + k];
    }
    if (tid < 32) { w1s[tid] = fc1_w[tid]; b1s[tid] = fc1_b[tid]; b2s[tid] = fc2_b[tid]; }
    __syncthreads();

    // ---------------- per-lane loop invariants ----------------
    // w1/b1 (f16, packed) in 16-bit A-fragment element order (K=32):
    // k(e) = (e>>3)*16 + hf*8 + (e&7)
    v16h w1A16, b1A16;
    #pragma unroll
    for (int e = 0; e < 16; ++e) {
        int f = ((e >> 3) << 4) + hf * 8 + (e & 7);
        w1A16[e] = (_Float16)w1s[f];
        b1A16[e] = (_Float16)b1s[f];
    }
    const v2h wl2 = {(_Float16)w1s[lane], (_Float16)w1s[lane]};   // h column n=lane
    const v2h bl2 = {(_Float16)b1s[lane], (_Float16)b1s[lane]};

    // fc2 bias as WMMA C-operand vectors (saves a v_add per element)
    v8f cbias[2];
    #pragma unroll
    for (int r = 0; r < 8; ++r) {
        cbias[0][r] = b2s[l15];
        cbias[1][r] = b2s[16 + l15];
    }

    // B-fragments of Wh (in-features 0..31) and Wa (32..63): B[k][n] = fc2_w[n][k]
    V16 WhB[2], WaB[2];
    #pragma unroll
    for (int nt = 0; nt < 2; ++nt) {
        int n = nt * 16 + l15;
        WhB[nt].h[0] = *(const v8h*)&w2t[n][hf * 8];
        WhB[nt].h[1] = *(const v8h*)&w2t[n][16 + hf * 8];
        WaB[nt].h[0] = *(const v8h*)&w2t[n][32 + hf * 8];
        WaB[nt].h[1] = *(const v8h*)&w2t[n][32 + 16 + hf * 8];
    }

    const v16h z16 = (v16h){0,0,0,0,0,0,0,0,0,0,0,0,0,0,0,0};
    const v2h  z2  = (v2h){0,0};
    const v8f  z8  = (v8f){0.f,0.f,0.f,0.f,0.f,0.f,0.f,0.f};

    v8f running[4][2];
    #pragma unroll
    for (int mt = 0; mt < 4; ++mt)
        #pragma unroll
        for (int nt = 0; nt < 2; ++nt)
            running[mt][nt] = z8;

    const float* xb = x + (size_t)b * S_TOTAL * 64;

    // ---------------- per-sample loop ----------------
    for (int si = 0; si < S_PER_WAVE; ++si) {
        const int s = s0 + si;

        // stage x row into LDS as packed f16
        float2 xv = *(const float2*)(xb + (size_t)s * 64 + lane * 2);
        v2h xp = {(_Float16)xv.x, (_Float16)xv.y};
        *(v2h*)&xh[wv][lane * 2] = xp;
        lds_fence();

        // h column-major f16 into LDS: lane owns feature n=lane (packed f16 math)
        #pragma unroll
        for (int k = 0; k < 64; k += 2) {
            v2h xk = *(const v2h*)&xh[wv][k];
            v2h hv = __builtin_elementwise_max(xk * wl2 + bl2, z2);
            *(v2h*)&hcm[wv][lane][k] = hv;
        }
        lds_fence();

        // B-fragments of h: B[k=var][n=feat]; k = kt*32 + (e>>3)*16 + hf*8 + (e&7)
        V16 hB[2][2];
        #pragma unroll
        for (int kt = 0; kt < 2; ++kt)
            #pragma unroll
            for (int nt = 0; nt < 2; ++nt) {
                int n = nt * 16 + l15;
                hB[kt][nt].h[0] = *(const v8h*)&hcm[wv][n][kt * 32 + hf * 8];
                hB[kt][nt].h[1] = *(const v8h*)&hcm[wv][n][kt * 32 + 16 + hf * 8];
            }

        // Phase 1: agg = adj_norm @ h, all 8 tiles accumulated first.
        // nt=0 / nt=1 chains interleave -> dependent WMMA pairs separated.
        v8f agga[4][2];
        #pragma unroll
        for (int mt = 0; mt < 4; ++mt) {
            int m = mt * 16 + l15;
            V16 a0, a1;
            a0.h[0] = *(const v8h*)&adjn16[m][hf * 8];
            a0.h[1] = *(const v8h*)&adjn16[m][16 + hf * 8];
            a1.h[0] = *(const v8h*)&adjn16[m][32 + hf * 8];
            a1.h[1] = *(const v8h*)&adjn16[m][48 + hf * 8];
            agga[mt][0] = wmma_f16(a0.v, hB[0][0].v, z8);
            agga[mt][1] = wmma_f16(a0.v, hB[0][1].v, z8);
            agga[mt][0] = wmma_f16(a1.v, hB[1][0].v, agga[mt][0]);
            agga[mt][1] = wmma_f16(a1.v, hB[1][1].v, agga[mt][1]);
        }

        // Phase 2: spill all tiles C-layout -> row-major f16 (A-layout reload)
        #pragma unroll
        for (int mt = 0; mt < 4; ++mt)
            #pragma unroll
            for (int nt = 0; nt < 2; ++nt) {
                int col = nt * 16 + l15;
                #pragma unroll
                for (int r = 0; r < 8; ++r)
                    aggrm[wv][mt * 16 + hf * 8 + r][col] = (_Float16)agga[mt][nt][r];
            }
        lds_fence();

        // Phase 3: h2 tiles relu(h@Wh + agg@Wa + b2) -> running sums
        #pragma unroll
        for (int mt = 0; mt < 4; ++mt) {
            int m = mt * 16 + l15;
            // h A-fragment built directly in registers: relu(x[m]*w1+b1), packed f16
            _Float16 xm = xh[wv][m];
            V16 hA;
            hA.v = __builtin_elementwise_max(w1A16 * xm + b1A16, z16);
            // agg A-fragment from LDS (two contiguous 16B runs)
            V16 aA;
            aA.h[0] = *(const v8h*)&aggrm[wv][m][hf * 8];
            aA.h[1] = *(const v8h*)&aggrm[wv][m][16 + hf * 8];
            v8f acc0 = wmma_f16(hA.v, WhB[0].v, cbias[0]);  // bias as C
            v8f acc1 = wmma_f16(hA.v, WhB[1].v, cbias[1]);
            acc0     = wmma_f16(aA.v, WaB[0].v, acc0);
            acc1     = wmma_f16(aA.v, WaB[1].v, acc1);
            running[mt][0] += __builtin_elementwise_max(acc0, z8);
            running[mt][1] += __builtin_elementwise_max(acc1, z8);
        }
    }

    // ---------------- merge partial means ----------------
    const float scale = 1.0f / (float)S_TOTAL;
    #pragma unroll
    for (int mt = 0; mt < 4; ++mt)
        #pragma unroll
        for (int nt = 0; nt < 2; ++nt) {
            int col = nt * 16 + l15;
            #pragma unroll
            for (int r = 0; r < 8; ++r) {
                int m = mt * 16 + hf * 8 + r;            // C-layout row
                unsafeAtomicAdd(&out[(size_t)b * 2048 + m * 32 + col],
                                running[mt][nt][r] * scale);
            }
        }
}

extern "C" void kernel_launch(void* const* d_in, const int* in_sizes, int n_in,
                              void* d_out, int out_size, void* d_ws, size_t ws_size,
                              hipStream_t stream) {
    const float* x     = (const float*)d_in[0];
    const float* adj   = (const float*)d_in[1];
    const float* fc1_w = (const float*)d_in[2];
    const float* fc1_b = (const float*)d_in[3];
    const float* fc2_w = (const float*)d_in[4];
    const float* fc2_b = (const float*)d_in[5];
    float* out = (float*)d_out;

    zero_out_kernel<<<dim3((out_size + 255) / 256), 256, 0, stream>>>(out, out_size);
    gnn_encoder_kernel<<<dim3(16, SCHUNKS), BLK, 0, stream>>>(
        x, adj, fc1_w, fc1_b, fc2_w, fc2_b, out);
}
